// GraphPolicyNetwork_51299089383461
// MI455X (gfx1250) — matrix operand order repaired and checked by reading, not compiled, not run on hardware
//
#include <hip/hip_runtime.h>
#include <hip/hip_bf16.h>

// ---------------------------------------------------------------------------
// Types for CDNA5 WMMA
// ---------------------------------------------------------------------------
typedef __attribute__((ext_vector_type(16))) __bf16 v16bf;
typedef __attribute__((ext_vector_type(8)))  float  v8f;

static __device__ __forceinline__ unsigned short f2bf(float f) {
    unsigned u = __float_as_uint(f);
    unsigned r = u + 0x7FFFu + ((u >> 16) & 1u);   // round-to-nearest-even
    return (unsigned short)(r >> 16);
}
static __device__ __forceinline__ void atomicMaxF(float* addr, float val) {
    if (val >= 0.0f) atomicMax((int*)addr, __float_as_int(val));
    else             atomicMin((unsigned int*)addr, __float_as_uint(val));
}

// Pack two 16-byte chunks into a 32-byte bf16 WMMA fragment.
union FragPack {
    uint4  q[2];
    v16bf  v;
};

// ---------------------------------------------------------------------------
// Generic utility kernels
// ---------------------------------------------------------------------------
__global__ void fill_f32_kernel(float* p, float v, int n) {
    int i = blockIdx.x * blockDim.x + threadIdx.x;
    if (i < n) p[i] = v;
}
// src[K][N] (f32) -> dst[N][K] (bf16), i.e. store weights transposed so the
// GEMM's B operand is K-contiguous per output column.
__global__ void conv_bf16_T_kernel(const float* __restrict__ s, unsigned short* __restrict__ d,
                                   int K, int Ncols) {
    int i = blockIdx.x * blockDim.x + threadIdx.x;
    if (i < K * Ncols) {
        int k = i / Ncols, n = i % Ncols;
        d[(size_t)n * K + k] = f2bf(s[i]);
    }
}

// ---------------------------------------------------------------------------
// bf16 WMMA GEMM: C[M,Ncols] = A[M,K](bf16,row-major) @ Bt[Ncols,K](bf16)^T
// (+bias)(+relu).  block = 256 threads = 8 waves; tile 128x128, K-step 32.
// wave w: rows [wr*32, wr*32+32), cols [wc*64, wc*64+64)  (wr=w&3, wc=w>>2)
// ---------------------------------------------------------------------------
#define BM 128
#define BN 128
#define BKK 32

__global__ __launch_bounds__(256) void gemm_bf16_wmma(
    const unsigned short* __restrict__ A, int lda,
    const unsigned short* __restrict__ Bt,          // [Ncols][K]
    const float* __restrict__ bias,
    float* __restrict__ Cf, unsigned short* __restrict__ Cb, int ldc,
    int M, int K, int relu)
{
    __shared__ unsigned short As[BM][40];    // rows x K-chunk, 80B row stride
    __shared__ unsigned short BsT[BN][40];   // cols x K-chunk, 80B row stride

    const int tid  = threadIdx.x;
    const int lane = tid & 31;
    const int w    = tid >> 5;
    const int wr   = w & 3;
    const int wc   = w >> 2;
    const int row0 = blockIdx.y * BM;
    const int col0 = blockIdx.x * BN;

    v8f acc[2][4];
    for (int i = 0; i < 2; ++i)
        for (int j = 0; j < 4; ++j)
            for (int r = 0; r < 8; ++r) acc[i][j][r] = 0.0f;

    const int ar = tid >> 1, ac = (tid & 1) * 16;   // A staging: 128 rows x 32 halves
    const int mhalf = lane >> 4;
    const int mrow  = lane & 15;

    for (int k0 = 0; k0 < K; k0 += BKK) {
        // ---- stage A tile (zero-fill out-of-range rows) ----
        uint4 a0 = {0, 0, 0, 0}, a1 = {0, 0, 0, 0};
        const int garow = row0 + ar;
        if (garow < M) {
            const uint4* gp = (const uint4*)(A + (size_t)garow * lda + k0 + ac);
            a0 = gp[0]; a1 = gp[1];
            if (k0 + BKK < K)
                __builtin_prefetch(A + (size_t)garow * lda + k0 + BKK + ac, 0, 0);
        }
        *(uint4*)&As[ar][ac]     = a0;
        *(uint4*)&As[ar][ac + 8] = a1;
        // ---- stage Bt tile (same shape as A: 128 cols x 32 halves of K) ----
        {
            const int col = ar;                       // reuse thread mapping
            const uint4* gp = (const uint4*)(Bt + (size_t)(col0 + col) * K + k0 + ac);
            uint4 b0 = gp[0], b1 = gp[1];
            if (k0 + BKK < K)
                __builtin_prefetch(Bt + (size_t)(col0 + col) * K + k0 + BKK + ac, 0, 0);
            *(uint4*)&BsT[col][ac]     = b0;
            *(uint4*)&BsT[col][ac + 8] = b1;
        }
        __syncthreads();

        // ---- A fragments (ISA 7.12.2): elems 0..7 = K 8*half..+7,
        //      elems 8..15 = K 16+8*half..+7  -> two contiguous b128 loads ----
        v16bf afr[2];
        for (int mi = 0; mi < 2; ++mi) {
            const int m = wr * 32 + mi * 16 + mrow;
            FragPack fp;
            fp.q[0] = *(const uint4*)&As[m][8 * mhalf];
            fp.q[1] = *(const uint4*)&As[m][16 + 8 * mhalf];
            afr[mi] = fp.v;
        }
        // ---- B fragments: lane holds column n, elems 0..7 = K 8*half..+7,
        //      elems 8..15 = K 16+8*half..+7 -> two b128 loads ----
        v16bf bfr[4];
        for (int ni = 0; ni < 4; ++ni) {
            const int n = wc * 64 + ni * 16 + mrow;
            FragPack fp;
            fp.q[0] = *(const uint4*)&BsT[n][8 * mhalf];
            fp.q[1] = *(const uint4*)&BsT[n][16 + 8 * mhalf];
            bfr[ni] = fp.v;
        }
        // ---- 8 WMMA per wave per K-step ----
        for (int mi = 0; mi < 2; ++mi)
            for (int ni = 0; ni < 4; ++ni)
                acc[mi][ni] = __builtin_amdgcn_wmma_f32_16x16x32_bf16(
                    false, afr[mi], false, bfr[ni],
                    (short)0, acc[mi][ni], false, false);
        __syncthreads();
    }

    // ---- epilogue: C layout — VGPR r holds M = r + 8*half, lane%16 = N ----
    for (int mi = 0; mi < 2; ++mi)
        for (int ni = 0; ni < 4; ++ni)
            for (int r = 0; r < 8; ++r) {
                const int grow = row0 + wr * 32 + mi * 16 + r + 8 * mhalf;
                const int gcol = col0 + wc * 64 + ni * 16 + mrow;
                if (grow < M) {
                    float v = acc[mi][ni][r];
                    if (bias) v += bias[gcol];
                    if (relu) v = v > 0.0f ? v : 0.0f;
                    if (Cf) Cf[(size_t)grow * ldc + gcol] = v;
                    if (Cb) Cb[(size_t)grow * ldc + gcol] = f2bf(v);
                }
            }
}

// ---------------------------------------------------------------------------
// Node encoder: h = LayerNorm(ReLU(x @ ne_w + ne_b)); grid=N, block=128
// ---------------------------------------------------------------------------
__global__ __launch_bounds__(128) void node_encode_kernel(
    const float* __restrict__ x, const float* __restrict__ w, const float* __restrict__ b,
    const float* __restrict__ g, const float* __restrict__ beta,
    float* __restrict__ h, unsigned short* __restrict__ hb)
{
    const int n = blockIdx.x, f = threadIdx.x;
    float acc = b[f];
    for (int k = 0; k < 6; ++k) acc += x[n * 6 + k] * w[k * 128 + f];
    acc = acc > 0.0f ? acc : 0.0f;

    __shared__ float red[128];
    red[f] = acc; __syncthreads();
    for (int s = 64; s > 0; s >>= 1) { if (f < s) red[f] += red[f + s]; __syncthreads(); }
    const float mu = red[0] * (1.0f / 128.0f); __syncthreads();
    const float d = acc - mu;
    red[f] = d * d; __syncthreads();
    for (int s = 64; s > 0; s >>= 1) { if (f < s) red[f] += red[f + s]; __syncthreads(); }
    const float var = red[0] * (1.0f / 128.0f);
    const float v = d * __frsqrt_rn(var + 1e-5f) * g[f] + beta[f];
    h[(size_t)n * 128 + f]  = v;
    hb[(size_t)n * 128 + f] = f2bf(v);
}

// ---------------------------------------------------------------------------
// Attention score dots: es[n,h]=feat[n,h,:].a_s[h], ed likewise. grid=N, block=32*H
// ---------------------------------------------------------------------------
__global__ void attn_scores_kernel(
    const float* __restrict__ feat, const float* __restrict__ a_s, const float* __restrict__ a_d,
    float* __restrict__ es, float* __restrict__ ed, int H)
{
    const int n = blockIdx.x;
    const int hh = threadIdx.x >> 5;
    const int lane = threadIdx.x & 31;
    float ss = 0.0f, sd = 0.0f;
    const size_t base = (size_t)n * H * 128 + hh * 128;
    for (int c = lane; c < 128; c += 32) {
        const float v = feat[base + c];
        ss += v * a_s[hh * 128 + c];
        sd += v * a_d[hh * 128 + c];
    }
    for (int off = 16; off; off >>= 1) {
        ss += __shfl_xor(ss, off, 32);
        sd += __shfl_xor(sd, off, 32);
    }
    if (lane == 0) { es[n * H + hh] = ss; ed[n * H + hh] = sd; }
}

// ---------------------------------------------------------------------------
// Attention scatter passes (edges + implicit self-loops at e >= E)
// ---------------------------------------------------------------------------
__global__ void edge_pass1_kernel(const int* __restrict__ ei, int E, int Etot,
                                  const float* __restrict__ es, const float* __restrict__ ed,
                                  int H, float* __restrict__ eo, float* __restrict__ mmax)
{
    const int e = blockIdx.x * blockDim.x + threadIdx.x;
    if (e >= Etot) return;
    int s, d;
    if (e < E) { s = ei[e]; d = ei[E + e]; } else { s = d = e - E; }
    for (int h = 0; h < H; ++h) {
        float v = es[s * H + h] + ed[d * H + h];
        v = v > 0.0f ? v : 0.2f * v;
        eo[(size_t)e * H + h] = v;
        atomicMaxF(&mmax[d * H + h], v);
    }
}

__global__ void edge_pass2_kernel(const int* __restrict__ ei, int E, int Etot,
                                  int H, float* __restrict__ eo,
                                  const float* __restrict__ mmax, float* __restrict__ den)
{
    const int e = blockIdx.x * blockDim.x + threadIdx.x;
    if (e >= Etot) return;
    int d;
    if (e < E) d = ei[E + e]; else d = e - E;
    for (int h = 0; h < H; ++h) {
        const float w = __expf(eo[(size_t)e * H + h] - mmax[d * H + h]);
        eo[(size_t)e * H + h] = w;
        atomicAdd(&den[d * H + h], w);
    }
}

__global__ __launch_bounds__(128) void edge_pass3_kernel(
    const int* __restrict__ ei, int E, int Etot, int H,
    const float* __restrict__ eo, const float* __restrict__ den,
    const float* __restrict__ feat, float* __restrict__ agg)
{
    const int e = blockIdx.x;
    const int c = threadIdx.x;
    int s, d;
    if (e < E) { s = ei[e]; d = ei[E + e]; } else { s = d = e - E; }
    for (int h = 0; h < H; ++h) {
        const float alpha = eo[(size_t)e * H + h] / (den[d * H + h] + 1e-16f);
        atomicAdd(&agg[(size_t)d * H * 128 + h * 128 + c],
                  alpha * feat[(size_t)s * H * 128 + h * 128 + c]);
    }
}

__global__ __launch_bounds__(128) void head_mean_kernel(
    const float* __restrict__ agg, const float* __restrict__ bias, int H, int relu,
    float* __restrict__ h, unsigned short* __restrict__ hb)
{
    const int n = blockIdx.x, c = threadIdx.x;
    float v = 0.0f;
    for (int hh = 0; hh < H; ++hh) v += agg[(size_t)n * H * 128 + hh * 128 + c];
    v = v / (float)H + bias[c];
    if (relu) v = v > 0.0f ? v : 0.0f;
    h[(size_t)n * 128 + c]  = v;
    hb[(size_t)n * 128 + c] = f2bf(v);
}

// ---------------------------------------------------------------------------
// Scorer: gather combined = [h[src] | h[dst] | relu(edge_attr@ee_w+ee_b)] (bf16)
// grid = chunk rows, block = 128
// ---------------------------------------------------------------------------
__global__ __launch_bounds__(128) void gather_combined_kernel(
    const float* __restrict__ h3, const int* __restrict__ ei, int E,
    int chunk0, const float* __restrict__ ea,
    const float* __restrict__ ee_w, const float* __restrict__ ee_b,
    unsigned short* __restrict__ comb)
{
    const int r = blockIdx.x;
    const int e = chunk0 + r;
    const int c = threadIdx.x;
    const int s = ei[e], d = ei[E + e];
    comb[(size_t)r * 320 + c]       = f2bf(h3[(size_t)s * 128 + c]);
    comb[(size_t)r * 320 + 128 + c] = f2bf(h3[(size_t)d * 128 + c]);
    if (c < 64) {
        float a = ee_b[c];
        for (int k = 0; k < 4; ++k) a += ea[(size_t)e * 4 + k] * ee_w[k * 64 + c];
        a = a > 0.0f ? a : 0.0f;
        comb[(size_t)r * 320 + 256 + c] = f2bf(a);
    }
}

__global__ void final_dot_kernel(const float* __restrict__ z2,
                                 const float* __restrict__ s3w, const float* __restrict__ s3b,
                                 float* __restrict__ out, int chunk0, int chunkM)
{
    const int r = blockIdx.x * blockDim.x + threadIdx.x;
    if (r >= chunkM) return;
    const float* z = z2 + (size_t)r * 128;
    float acc = s3b[0];
    for (int c = 0; c < 128; ++c) acc += z[c] * s3w[c];
    out[chunk0 + r] = acc;
}

// ---------------------------------------------------------------------------
// Host launcher
// ---------------------------------------------------------------------------
static inline char* ws_bump(char*& p, size_t bytes) {
    char* r = p;
    p += (bytes + 255) & ~(size_t)255;
    return r;
}

extern "C" void kernel_launch(void* const* d_in, const int* in_sizes, int n_in,
                              void* d_out, int out_size, void* d_ws, size_t ws_size,
                              hipStream_t stream)
{
    const int N = 20000, E = 320000, ETOT = E + N;
    const int CHUNK = 16000;                 // 20 chunks, each 125 x 128 rows

    const float* x    = (const float*)d_in[0];
    const int*   ei   = (const int*)  d_in[1];
    const float* ea   = (const float*)d_in[2];
    const float* ne_w = (const float*)d_in[3];
    const float* ne_b = (const float*)d_in[4];
    const float* ln_g = (const float*)d_in[5];
    const float* ln_b = (const float*)d_in[6];
    const float* ee_w = (const float*)d_in[7];
    const float* ee_b = (const float*)d_in[8];
    const float* gW[3]  = { (const float*)d_in[9],  (const float*)d_in[13], (const float*)d_in[17] };
    const float* gAS[3] = { (const float*)d_in[10], (const float*)d_in[14], (const float*)d_in[18] };
    const float* gAD[3] = { (const float*)d_in[11], (const float*)d_in[15], (const float*)d_in[19] };
    const float* gB[3]  = { (const float*)d_in[12], (const float*)d_in[16], (const float*)d_in[20] };
    const float* s1_w = (const float*)d_in[21];
    const float* s1_b = (const float*)d_in[22];
    const float* s2_w = (const float*)d_in[23];
    const float* s2_b = (const float*)d_in[24];
    const float* s3_w = (const float*)d_in[25];
    const float* s3_b = (const float*)d_in[26];
    float* out = (float*)d_out;

    // ---- workspace layout ----
    char* wp = (char*)d_ws;
    unsigned short* hb   = (unsigned short*)ws_bump(wp, (size_t)N * 128 * 2);
    float*          h    = (float*)         ws_bump(wp, (size_t)N * 128 * 4);
    float*          feat = (float*)         ws_bump(wp, (size_t)N * 512 * 4);
    float*          agg  = (float*)         ws_bump(wp, (size_t)N * 512 * 4);
    float*          es   = (float*)         ws_bump(wp, (size_t)N * 4 * 4);
    float*          ed   = (float*)         ws_bump(wp, (size_t)N * 4 * 4);
    float*          eo   = (float*)         ws_bump(wp, (size_t)ETOT * 4 * 4);
    float*          mmax = (float*)         ws_bump(wp, (size_t)N * 4 * 4);
    float*          den  = (float*)         ws_bump(wp, (size_t)N * 4 * 4);
    unsigned short* gWbT[3];                                  // transposed bf16 weights
    gWbT[0] = (unsigned short*)ws_bump(wp, (size_t)128 * 512 * 2);
    gWbT[1] = (unsigned short*)ws_bump(wp, (size_t)128 * 512 * 2);
    gWbT[2] = (unsigned short*)ws_bump(wp, (size_t)128 * 256 * 2);
    unsigned short* s1_wbT = (unsigned short*)ws_bump(wp, (size_t)320 * 256 * 2);
    unsigned short* s2_wbT = (unsigned short*)ws_bump(wp, (size_t)256 * 128 * 2);
    unsigned short* comb   = (unsigned short*)ws_bump(wp, (size_t)CHUNK * 320 * 2);
    unsigned short* z1b    = (unsigned short*)ws_bump(wp, (size_t)CHUNK * 256 * 2);
    float*          z2     = (float*)         ws_bump(wp, (size_t)CHUNK * 128 * 4);
    (void)ws_size; (void)n_in; (void)in_sizes; (void)out_size;

    // ---- weight conversions to transposed bf16  (src[K][N] -> dst[N][K]) ----
    {
        const int Ks[5]       = { 128, 128, 128, 320, 256 };
        const int Ns[5]       = { 512, 512, 256, 256, 128 };
        const float* src[5]   = { gW[0], gW[1], gW[2], s1_w, s2_w };
        unsigned short* dst[5] = { gWbT[0], gWbT[1], gWbT[2], s1_wbT, s2_wbT };
        for (int i = 0; i < 5; ++i) {
            const int cnt = Ks[i] * Ns[i];
            conv_bf16_T_kernel<<<(cnt + 255) / 256, 256, 0, stream>>>(src[i], dst[i], Ks[i], Ns[i]);
        }
    }

    // ---- node encoder ----
    node_encode_kernel<<<N, 128, 0, stream>>>(x, ne_w, ne_b, ln_g, ln_b, h, hb);

    // ---- 3 GAT layers ----
    const int Hh[3] = { 4, 4, 2 };
    const int NC[3] = { 512, 512, 256 };
    for (int l = 0; l < 3; ++l) {
        const int H = Hh[l], Ncols = NC[l];
        // feat = hb @ W  (bf16 WMMA, f32 out)
        dim3 grid(Ncols / BN, (N + BM - 1) / BM);
        gemm_bf16_wmma<<<grid, 256, 0, stream>>>(hb, 128, gWbT[l], nullptr,
                                                 feat, nullptr, Ncols, N, 128, 0);
        attn_scores_kernel<<<N, 32 * H, 0, stream>>>(feat, gAS[l], gAD[l], es, ed, H);
        fill_f32_kernel<<<(N * H + 255) / 256, 256, 0, stream>>>(mmax, -__builtin_inff(), N * H);
        fill_f32_kernel<<<(N * H + 255) / 256, 256, 0, stream>>>(den, 0.0f, N * H);
        fill_f32_kernel<<<((size_t)N * Ncols + 255) / 256, 256, 0, stream>>>(agg, 0.0f, N * Ncols);
        edge_pass1_kernel<<<(ETOT + 255) / 256, 256, 0, stream>>>(ei, E, ETOT, es, ed, H, eo, mmax);
        edge_pass2_kernel<<<(ETOT + 255) / 256, 256, 0, stream>>>(ei, E, ETOT, H, eo, mmax, den);
        edge_pass3_kernel<<<ETOT, 128, 0, stream>>>(ei, E, ETOT, H, eo, den, feat, agg);
        head_mean_kernel<<<N, 128, 0, stream>>>(agg, gB[l], H, (l < 2) ? 1 : 0, h, hb);
    }

    // ---- edge scorer MLP, chunked ----
    for (int c0 = 0; c0 < E; c0 += CHUNK) {
        gather_combined_kernel<<<CHUNK, 128, 0, stream>>>(h, ei, E, c0, ea, ee_w, ee_b, comb);
        // z1b = relu(comb @ s1_w + s1_b)  -> bf16
        dim3 g1(256 / BN, CHUNK / BM);
        gemm_bf16_wmma<<<g1, 256, 0, stream>>>(comb, 320, s1_wbT, s1_b,
                                               nullptr, z1b, 256, CHUNK, 320, 1);
        // z2 = relu(z1b @ s2_w + s2_b)  -> f32
        dim3 g2(128 / BN, CHUNK / BM);
        gemm_bf16_wmma<<<g2, 256, 0, stream>>>(z1b, 256, s2_wbT, s2_b,
                                               z2, nullptr, 128, CHUNK, 256, 1);
        final_dot_kernel<<<(CHUNK + 255) / 256, 256, 0, stream>>>(z2, s3_w, s3_b, out, c0, CHUNK);
    }
}
